// AttentionLocalMessageFunction_53910429499700
// MI455X (gfx1250) — matrix operand order repaired and checked by reading, not compiled, not run on hardware
//
#include <hip/hip_runtime.h>
#include <hip/hip_bf16.h>

typedef __attribute__((ext_vector_type(16))) _Float16 v16h;
typedef __attribute__((ext_vector_type(8)))  float    v8f;

#define WMMA_F16(A, B, C) \
  __builtin_amdgcn_wmma_f32_16x16x32_f16(false, (A), false, (B), (short)0, (C), false, false)

namespace almf {

constexpr int kN = 50000;
constexpr int kE = 800000;
constexpr int kETiles = kE / 16;   // 50000
constexpr int kNTiles = kN / 16;   // 3125

// Workspace layout (bytes). All offsets multiples of 32 (v16h alignment).
constexpr size_t OFF_W1CS = 0;            // f16 [32 ct][3 kc][32 lane][16]  = 98304 B  (layer1 combined, B-swizzled)
constexpr size_t OFF_B1C  = 98304;        // f32 [512]                       = 2048 B
constexpr size_t OFF_W2VS = 100352;       // f16 [8 hp][32 lane][16]         = 8192 B   (value layer2, B-swizzled)
constexpr size_t OFF_B2V  = 108544;       // f32 [8][16]                     = 512 B
constexpr size_t OFF_SW2  = 109056;       // f32 [8][32]                     = 1024 B
constexpr size_t OFF_SB2  = 110080;       // f32 [8]                         = 32 B
constexpr size_t OFF_PW1S = 110336;       // f16 [16 ct][2 kc][32][16]       = 32768 B  (psi layer1, B-swizzled)
constexpr size_t OFF_PB1  = 143104;       // f32 [256]                       = 1024 B
constexpr size_t OFF_PW2S = 144128;       // f16 [8 ct][8 kc][32][16]        = 65536 B  (psi layer2, B-swizzled)
constexpr size_t OFF_PB2  = 209664;       // f32 [128]                       = 512 B
constexpr size_t OFF_NUM  = 210176;       // f32 [N][64]                     = 12.8 MB
constexpr size_t OFF_DEN  = 13010176;     // f32 [N][4]                      = 0.8 MB
// total ~13.2 MB

// ---- A-fragment loader: 16-bit A matrix 16x32 (M x K), documented VGPR layout.
// lane L: M = L%16, g = L/16. half t (=2v+hl): K = k0 + 8g + (v>=4?16:0) + 2*(v&3) + hl
__device__ __forceinline__ v16h load_a16(const _Float16* __restrict__ base, int ldk,
                                         int k0, int lane) {
  const int m = lane & 15, g = lane >> 4;
  v16h a;
#pragma unroll
  for (int t = 0; t < 16; ++t) {
    const int v = t >> 1, hl = t & 1;
    const int k = k0 + 8 * g + ((v & 4) ? 16 : 0) + 2 * (v & 3) + hl;
    a[t] = base[m * ldk + k];
  }
  return a;
}

// B fragment swizzle index (used by the pack kernel): 16-bit B matrix 32x16 (K x N).
// lane L: N = L%16, g = L/16. half t (=2v+hl): K = k0 + 8g + (v>=4?16:0) + 2*(v&3) + hl
__device__ __forceinline__ void b_swizzle_kn(int lane, int t, int& k_local, int& n_local) {
  const int g = lane >> 4, v = t >> 1, hl = t & 1;
  k_local = 8 * g + ((v & 4) ? 16 : 0) + 2 * (v & 3) + hl;
  n_local = lane & 15;
}

// ------------------------------------------------------------------ zero
__global__ __launch_bounds__(256) void zero_kernel(float* __restrict__ num,
                                                   float* __restrict__ den) {
  const int tid = blockIdx.x * blockDim.x + threadIdx.x;
  const int stride = gridDim.x * blockDim.x;
  for (int i = tid; i < kN * 64; i += stride) num[i] = 0.0f;
  for (int i = tid; i < kN * 4;  i += stride) den[i] = 0.0f;
}

// ------------------------------------------------------------------ pack
__global__ __launch_bounds__(256) void pack_kernel(
    const float* __restrict__ vW1, const float* __restrict__ vb1,
    const float* __restrict__ vW2, const float* __restrict__ vb2,
    const float* __restrict__ sW1, const float* __restrict__ sb1,
    const float* __restrict__ sW2, const float* __restrict__ sb2,
    const float* __restrict__ pW1, const float* __restrict__ pb1,
    const float* __restrict__ pW2, const float* __restrict__ pb2,
    char* __restrict__ ws) {
  _Float16* W1cs = (_Float16*)(ws + OFF_W1CS);
  float*    b1c  = (float*)(ws + OFF_B1C);
  _Float16* W2vs = (_Float16*)(ws + OFF_W2VS);
  float*    b2v  = (float*)(ws + OFF_B2V);
  float*    sw2  = (float*)(ws + OFF_SW2);
  float*    sb2o = (float*)(ws + OFF_SB2);
  _Float16* pW1s = (_Float16*)(ws + OFF_PW1S);
  float*    pb1o = (float*)(ws + OFF_PB1);
  _Float16* pW2s = (_Float16*)(ws + OFF_PW2S);
  float*    pb2o = (float*)(ws + OFF_PB2);

  const int tid = blockIdx.x * blockDim.x + threadIdx.x;
  const int stride = gridDim.x * blockDim.x;

  // Layer-1 combined weights [K=80 pad 96][col 0..255 value | 256..511 score], B-swizzled.
  for (int i = tid; i < 32 * 3 * 32 * 16; i += stride) {
    const int t = i & 15, lane = (i >> 4) & 31, rest = i >> 9;
    const int kc = rest % 3, ct = rest / 3;
    int kl, nl; b_swizzle_kn(lane, t, kl, nl);
    const int k = kc * 32 + kl, col = ct * 16 + nl;
    float val = 0.0f;
    if (k < 80) {
      if (col < 256) val = vW1[((col >> 5) * 80 + k) * 32 + (col & 31)];
      else { const int c2 = col - 256; val = sW1[((c2 >> 5) * 80 + k) * 32 + (c2 & 31)]; }
    }
    W1cs[i] = (_Float16)val;
  }
  for (int i = tid; i < 512; i += stride) b1c[i] = (i < 256) ? vb1[i] : sb1[i - 256];

  // Value layer-2 [8 hp][K=32][N=16], B-swizzled.
  for (int i = tid; i < 8 * 32 * 16; i += stride) {
    const int t = i & 15, lane = (i >> 4) & 31, hp = i >> 9;
    int kl, nl; b_swizzle_kn(lane, t, kl, nl);
    W2vs[i] = (_Float16)vW2[(hp * 32 + kl) * 16 + nl];
  }
  for (int i = tid; i < 128; i += stride) b2v[i]  = vb2[i];
  for (int i = tid; i < 256; i += stride) sw2[i]  = sW2[i];
  for (int i = tid; i < 8;   i += stride) sb2o[i] = sb2[i];

  // Psi layer-1 [K=64][N=256], B-swizzled.
  for (int i = tid; i < 16 * 2 * 32 * 16; i += stride) {
    const int t = i & 15, lane = (i >> 4) & 31, rest = i >> 9;
    const int kc = rest & 1, ct = rest >> 1;
    int kl, nl; b_swizzle_kn(lane, t, kl, nl);
    pW1s[i] = (_Float16)pW1[(kc * 32 + kl) * 256 + ct * 16 + nl];
  }
  for (int i = tid; i < 256; i += stride) pb1o[i] = pb1[i];

  // Psi layer-2 [K=256][N=128], B-swizzled.
  for (int i = tid; i < 8 * 8 * 32 * 16; i += stride) {
    const int t = i & 15, lane = (i >> 4) & 31, rest = i >> 9;
    const int kc = rest & 7, ct = rest >> 3;
    int kl, nl; b_swizzle_kn(lane, t, kl, nl);
    pW2s[i] = (_Float16)pW2[(kc * 32 + kl) * 128 + ct * 16 + nl];
  }
  for (int i = tid; i < 128; i += stride) pb2o[i] = pb2[i];
}

// ------------------------------------------------------------------ edges
// One wave = one 16-edge tile. Layer1: [16x96]x[96x512] = 3 WMMA per 16-col tile.
__global__ __launch_bounds__(64) void edge_kernel(
    const float* __restrict__ coords, const float* __restrict__ ef,
    const int* __restrict__ src, const int* __restrict__ dst,
    const float* __restrict__ nf,
    const v16h* __restrict__ W1cs, const float* __restrict__ b1c,
    const v16h* __restrict__ W2vs, const float* __restrict__ b2v,
    const float* __restrict__ sW2, const float* __restrict__ sb2,
    float* __restrict__ num, float* __restrict__ den) {
  __shared__ __align__(32) _Float16 sX[2][16 * 96];
  __shared__ __align__(32) _Float16 sH[2][16 * 512];
  __shared__ float sEW[2][16 * 8];
  __shared__ int   sSrc[2][16];
  __shared__ int   sDst[2][16];
  __shared__ float sNF[2][16];

  const int lane = threadIdx.x & 31;
  const int w    = threadIdx.x >> 5;
  const int tile = blockIdx.x * 2 + w;
  if (tile >= kETiles) return;
  const int ebase = tile * 16;

  if (ebase + 16 < kE) __builtin_prefetch(&ef[(size_t)(ebase + 16) * 16], 0, 1);

  if (lane < 16) {
    sSrc[w][lane] = src[ebase + lane];
    sDst[w][lane] = dst[ebase + lane];
    sNF [w][lane] = nf [ebase + lane];
  }
  __builtin_amdgcn_wave_barrier();

  // Gather x = [edge_feat(16) | coord[src](32) | coord[dst](32) | pad(16)] as f16.
  for (int idx = lane; idx < 16 * 96; idx += 32) {
    const int e = idx / 96, k = idx - e * 96;
    float xv = 0.0f;
    if (k < 16)      xv = ef[(size_t)(ebase + e) * 16 + k];
    else if (k < 48) xv = coords[(size_t)sSrc[w][e] * 32 + (k - 16)];
    else if (k < 80) xv = coords[(size_t)sDst[w][e] * 32 + (k - 48)];
    sX[w][e * 96 + k] = (_Float16)xv;
  }
  __builtin_amdgcn_wave_barrier();

  const int n_lane = lane & 15, g = lane >> 4;

  // Layer 1: A frags are ct-invariant; B frags are single contiguous v16h loads.
  const v16h a0 = load_a16(&sX[w][0], 96, 0,  lane);
  const v16h a1 = load_a16(&sX[w][0], 96, 32, lane);
  const v16h a2 = load_a16(&sX[w][0], 96, 64, lane);

  for (int ct = 0; ct < 32; ++ct) {
    const float bias = b1c[ct * 16 + n_lane];
    v8f c;
#pragma unroll
    for (int v = 0; v < 8; ++v) c[v] = bias;
    c = WMMA_F16(a0, W1cs[(ct * 3 + 0) * 32 + lane], c);
    c = WMMA_F16(a1, W1cs[(ct * 3 + 1) * 32 + lane], c);
    c = WMMA_F16(a2, W1cs[(ct * 3 + 2) * 32 + lane], c);
#pragma unroll
    for (int v = 0; v < 8; ++v) {
      const float r = c[v] > 0.0f ? c[v] : 0.0f;            // ReLU
      sH[w][(v + 8 * g) * 512 + ct * 16 + n_lane] = (_Float16)r;
    }
  }
  __builtin_amdgcn_wave_barrier();

  // Scores: 128 (edge,hp) dots of length 32 over hs (cols 256..511), then exp + den atomics.
  for (int it = 0; it < 4; ++it) {
    const int idx = it * 32 + lane;
    const int e = idx >> 3, hp = idx & 7;
    const v16h* hrow = (const v16h*)&sH[w][e * 512 + 256 + hp * 32];
    const v16h h0 = hrow[0], h1 = hrow[1];
    const float* wv = &sW2[hp * 32];
    float s = sb2[hp];
#pragma unroll
    for (int d = 0; d < 16; ++d) s += (float)h0[d] * wv[d];
#pragma unroll
    for (int d = 0; d < 16; ++d) s += (float)h1[d] * wv[16 + d];
    s *= sNF[w][e];                                         // mask score before exp
    const float ew = __expf(s);
    sEW[w][e * 8 + hp] = ew;
    const int node = (hp & 1) ? sDst[w][e] : sSrc[w][e];
    atomicAdd(&den[(size_t)node * 4 + (hp >> 1)], ew);      // den gets raw exp
  }
  __builtin_amdgcn_wave_barrier();

  // Layer 2 values: one WMMA per (head,port); scatter straight from D fragment.
  for (int hp = 0; hp < 8; ++hp) {
    const v16h aV = load_a16(&sH[w][0], 512, hp * 32, lane);
    const v16h bV = W2vs[hp * 32 + lane];
    const float bias = b2v[hp * 16 + n_lane];
    v8f c;
#pragma unroll
    for (int v = 0; v < 8; ++v) c[v] = bias;
    c = WMMA_F16(aV, bV, c);
    const int h = hp >> 1;
#pragma unroll
    for (int v = 0; v < 8; ++v) {
      const int m = v + 8 * g;                              // edge row in tile
      const float wgt = sEW[w][m * 8 + hp] * sNF[w][m];     // exp * non_fictitious
      const int node = (hp & 1) ? sDst[w][m] : sSrc[w][m];
      atomicAdd(&num[(size_t)node * 64 + h * 16 + n_lane], c[v] * wgt);
    }
  }
}

// ------------------------------------------------------------------ nodes
// One wave = one 16-node tile: normalize, psi MLP 64->256->128.
__global__ __launch_bounds__(64) void node_kernel(
    const float* __restrict__ num, const float* __restrict__ den,
    const float* __restrict__ nfa,
    const v16h* __restrict__ pW1s, const float* __restrict__ pb1,
    const v16h* __restrict__ pW2s, const float* __restrict__ pb2,
    float* __restrict__ out) {
  __shared__ __align__(32) _Float16 sV[2][16 * 64];
  __shared__ __align__(32) _Float16 sH2[2][16 * 256];

  const int lane = threadIdx.x & 31;
  const int w    = threadIdx.x >> 5;
  const int tile = blockIdx.x * 2 + w;
  if (tile >= kNTiles) return;
  const int nbase = tile * 16;

  // v' = (num * nfa) / (den * nfa + eps)
  for (int idx = lane; idx < 16 * 64; idx += 32) {
    const int n = idx >> 6, c = idx & 63;
    const size_t node = (size_t)(nbase + n);
    const float a  = nfa[node];
    const float nm = num[node * 64 + c] * a;
    const float dn = den[node * 4 + (c >> 4)] * a + 1e-9f;
    sV[w][n * 64 + c] = (_Float16)(nm / dn);
  }
  __builtin_amdgcn_wave_barrier();

  const int n_lane = lane & 15, g = lane >> 4;

  const v16h a0 = load_a16(&sV[w][0], 64, 0,  lane);
  const v16h a1 = load_a16(&sV[w][0], 64, 32, lane);
  for (int ct = 0; ct < 16; ++ct) {
    const float bias = pb1[ct * 16 + n_lane];
    v8f c;
#pragma unroll
    for (int v = 0; v < 8; ++v) c[v] = bias;
    c = WMMA_F16(a0, pW1s[(ct * 2 + 0) * 32 + lane], c);
    c = WMMA_F16(a1, pW1s[(ct * 2 + 1) * 32 + lane], c);
#pragma unroll
    for (int v = 0; v < 8; ++v) {
      const float r = c[v] > 0.0f ? c[v] : 0.0f;
      sH2[w][(v + 8 * g) * 256 + ct * 16 + n_lane] = (_Float16)r;
    }
  }
  __builtin_amdgcn_wave_barrier();

  // Preload the 8 A-fragments of h (K=256) and the per-row output masks.
  v16h aF[8];
#pragma unroll
  for (int kc = 0; kc < 8; ++kc) aF[kc] = load_a16(&sH2[w][0], 256, kc * 32, lane);
  float fa[8];
#pragma unroll
  for (int v = 0; v < 8; ++v) fa[v] = nfa[nbase + v + 8 * g];

  for (int ct = 0; ct < 8; ++ct) {
    const float bias = pb2[ct * 16 + n_lane];
    v8f c;
#pragma unroll
    for (int v = 0; v < 8; ++v) c[v] = bias;
#pragma unroll
    for (int kc = 0; kc < 8; ++kc) c = WMMA_F16(aF[kc], pW2s[(ct * 8 + kc) * 32 + lane], c);
#pragma unroll
    for (int v = 0; v < 8; ++v) {
      const size_t node = (size_t)(nbase + v + 8 * g);
      out[node * 128 + ct * 16 + n_lane] = c[v] * fa[v];
    }
  }
}

}  // namespace almf

extern "C" void kernel_launch(void* const* d_in, const int* in_sizes, int n_in,
                              void* d_out, int out_size, void* d_ws, size_t ws_size,
                              hipStream_t stream) {
  (void)in_sizes; (void)n_in; (void)out_size; (void)ws_size;
  using namespace almf;

  const float* coords = (const float*)d_in[0];
  const float* ef     = (const float*)d_in[1];
  const int*   src    = (const int*)d_in[2];
  const int*   dst    = (const int*)d_in[3];
  const float* nf     = (const float*)d_in[4];
  const float* nfa    = (const float*)d_in[5];
  const float* vW1 = (const float*)d_in[6];
  const float* vb1 = (const float*)d_in[7];
  const float* vW2 = (const float*)d_in[8];
  const float* vb2 = (const float*)d_in[9];
  const float* sW1 = (const float*)d_in[10];
  const float* sb1 = (const float*)d_in[11];
  const float* sW2 = (const float*)d_in[12];
  const float* sb2 = (const float*)d_in[13];
  const float* pW1 = (const float*)d_in[14];
  const float* pb1 = (const float*)d_in[15];
  const float* pW2 = (const float*)d_in[16];
  const float* pb2 = (const float*)d_in[17];

  char* ws = (char*)d_ws;
  float* num = (float*)(ws + OFF_NUM);
  float* den = (float*)(ws + OFF_DEN);
  float* out = (float*)d_out;

  hipLaunchKernelGGL(zero_kernel, dim3(2048), dim3(256), 0, stream, num, den);
  hipLaunchKernelGGL(pack_kernel, dim3(192), dim3(256), 0, stream,
                     vW1, vb1, vW2, vb2, sW1, sb1, sW2, sb2, pW1, pb1, pW2, pb2, ws);
  hipLaunchKernelGGL(edge_kernel, dim3(kETiles / 2), dim3(64), 0, stream,
                     coords, ef, src, dst, nf,
                     (const v16h*)(ws + OFF_W1CS), (const float*)(ws + OFF_B1C),
                     (const v16h*)(ws + OFF_W2VS), (const float*)(ws + OFF_B2V),
                     (const float*)(ws + OFF_SW2), (const float*)(ws + OFF_SB2),
                     num, den);
  hipLaunchKernelGGL(node_kernel, dim3((kNTiles + 1) / 2), dim3(64), 0, stream,
                     num, den, nfa,
                     (const v16h*)(ws + OFF_PW1S), (const float*)(ws + OFF_PB1),
                     (const v16h*)(ws + OFF_PW2S), (const float*)(ws + OFF_PB2),
                     out);
}
